// KAN_Layer_88098369175698
// MI455X (gfx1250) — matrix operand order repaired and checked by reading, not compiled, not run on hardware
//
#include <hip/hip_runtime.h>
#include <hip/hip_bf16.h>

// KAN layer as f16 WMMA GEMM (CDNA5 / gfx1250):
//   pass 1 : A[8192][4096] (f16)  = bspline_basis(tanh(x))        -> d_ws
//   pass 1b: Bw[512][4096] (f16)  = f16(spline_coeffs)            -> d_ws + 64MB
//   pass 2 : C[8192][512] (f32)   = A x Bw^T
//            double-buffered LDS fed by global_load_async_to_lds_b128
//            (ASYNCcnt), 8x v_wmma_f32_16x16x32_f16 per wave per K-step.
//            K-loop kept at unroll 1 with a dynamic stage offset so the
//            accumulators keep a single register home (in-place WMMA).
// Workspace requirement: 8192*4096*2 + 512*4096*2 = 71,303,168 bytes (~68 MB).

typedef __attribute__((ext_vector_type(16))) _Float16 v16h;
typedef __attribute__((ext_vector_type(8)))  _Float16 v8h;
typedef __attribute__((ext_vector_type(8)))  float    v8f;

__device__ __forceinline__ v16h concat16(v8h lo, v8h hi) {
    return __builtin_shufflevector(lo, hi, 0, 1, 2, 3, 4, 5, 6, 7,
                                           8, 9, 10, 11, 12, 13, 14, 15);
}

// ---------------------------------------------------------------------------
// CDNA5 async copy: global -> LDS, 16 bytes per lane, tracked by ASYNCcnt.
// lds_off = low 32 bits of the generic shared-memory address (ISA: the LDS
// aperture maps addr[31:0] directly to the LDS byte offset).
// ---------------------------------------------------------------------------
__device__ __forceinline__ void async_copy_b128(uint32_t lds_off, const void* g) {
    asm volatile("global_load_async_to_lds_b128 %0, %1, off"
                 :: "v"(lds_off), "v"((unsigned long long)(uintptr_t)g)
                 : "memory");
}
__device__ __forceinline__ void wait_async0() {
    asm volatile("s_wait_asynccnt 0x0" ::: "memory");
}

// ---------------------------------------------------------------------------
// Cox-de Boor, order 3, GRID_SIZE=5 -> 12 uniform knots on [-1,1], h = 2/11.
// Uniform spacing => all (dl, dr) = k*h > EPS, the jnp.where guards vanish.
// ---------------------------------------------------------------------------
__device__ __forceinline__ void bspline8(float xn, float out[8]) {
    float g[12];
#pragma unroll
    for (int j = 0; j < 12; ++j) g[j] = -1.0f + (2.0f / 11.0f) * (float)j;

    float B0[11];
#pragma unroll
    for (int j = 0; j < 11; ++j)
        B0[j] = (xn >= g[j] && xn < g[j + 1]) ? 1.0f : 0.0f;

    const float inv1 = 11.0f / 2.0f;
    float B1[10];
#pragma unroll
    for (int j = 0; j < 10; ++j)
        B1[j] = (xn - g[j]) * inv1 * B0[j] + (g[j + 2] - xn) * inv1 * B0[j + 1];

    const float inv2 = 11.0f / 4.0f;
    float B2[9];
#pragma unroll
    for (int j = 0; j < 9; ++j)
        B2[j] = (xn - g[j]) * inv2 * B1[j] + (g[j + 3] - xn) * inv2 * B1[j + 1];

    const float inv3 = 11.0f / 6.0f;
#pragma unroll
    for (int j = 0; j < 8; ++j)
        out[j] = (xn - g[j]) * inv3 * B2[j] + (g[j + 4] - xn) * inv3 * B2[j + 1];
}

// ---------------------------------------------------------------------------
// Pass 1: one thread per x element; 8 f16 basis values out (16B, coalesced).
// ---------------------------------------------------------------------------
__global__ __launch_bounds__(256) void kan_basis_f16(const float* __restrict__ x,
                                                     _Float16* __restrict__ A) {
    const int tid = blockIdx.x * 256 + threadIdx.x;
    const float xn = tanhf(x[tid]);
    float b[8];
    bspline8(xn, b);
    union { _Float16 h[8]; uint4 u; } pk;
#pragma unroll
    for (int j = 0; j < 8; ++j) pk.h[j] = (_Float16)b[j];
    reinterpret_cast<uint4*>(A)[tid] = pk.u;
}

// ---------------------------------------------------------------------------
// Pass 1b: f32 -> f16 convert of coeffs; W[o][i][k] is already K-major.
// ---------------------------------------------------------------------------
__global__ __launch_bounds__(256) void kan_coeff_f16(const float* __restrict__ W,
                                                     _Float16* __restrict__ Bw) {
    const int tid = blockIdx.x * 256 + threadIdx.x;
    const float4* src = reinterpret_cast<const float4*>(W) + (size_t)tid * 2;
    const float4 a = src[0];
    const float4 b = src[1];
    union { _Float16 h[8]; uint4 u; } pk;
    pk.h[0] = (_Float16)a.x; pk.h[1] = (_Float16)a.y;
    pk.h[2] = (_Float16)a.z; pk.h[3] = (_Float16)a.w;
    pk.h[4] = (_Float16)b.x; pk.h[5] = (_Float16)b.y;
    pk.h[6] = (_Float16)b.z; pk.h[7] = (_Float16)b.w;
    reinterpret_cast<uint4*>(Bw)[tid] = pk.u;
}

// ---------------------------------------------------------------------------
// Pass 2: tiled WMMA GEMM, double-buffered async LDS staging.
//   block: 256 threads = 8 wave32; tile BM=128 x BN=128; K-step 32.
//   waves: 4 row-groups x 2 col-groups; each wave = 32x64 = 2x4 accumulators.
// ---------------------------------------------------------------------------
#define GEMM_K    4096
#define GEMM_N    512
#define BM        128
#define BN        128
#define BK        32
#define LDS_PITCH 40          // halves; 80B = 20 banks -> conflict-free b128
#define KSTEPS    (GEMM_K / BK)
#define STAGE_H   (BM * LDS_PITCH)      // halves per stage (A == B size)

__global__ __launch_bounds__(256) void kan_wmma_gemm(const _Float16* __restrict__ A,
                                                     const _Float16* __restrict__ Bw,
                                                     float* __restrict__ C) {
    __shared__ __align__(16) _Float16 As[2 * STAGE_H];
    __shared__ __align__(16) _Float16 Bs[2 * STAGE_H];

    const int tid  = threadIdx.x;
    const int lane = tid & 31;
    const int wid  = tid >> 5;
    const int m0   = blockIdx.y * BM;
    const int n0   = blockIdx.x * BN;

    const int wm = (wid & 3) * 32;      // wave row offset inside tile
    const int wn = (wid >> 2) * 64;     // wave col offset inside tile

    v8f acc[2][4] = {};

    const int lm  = lane & 15;
    const int kfA = (lane < 16) ? 0 : 8;    // A: K {0..7,16..23} | {8..15,24..31}
    const int kfB = (lane < 16) ? 0 : 8;    // B: K {0..7,16..23} | {8..15,24..31}

    // --- per-thread staging coordinates: row = tid>>1, 32B chunk = tid&1 ---
    const int r_st = tid >> 1;              // 0..127
    const int c_st = (tid & 1) * 16;        // halves (two b128 per thread)
    const _Float16* gA = A  + (size_t)(m0 + r_st) * GEMM_K + c_st;
    const _Float16* gB = Bw + (size_t)(n0 + r_st) * GEMM_K + c_st;

    // async LDS destinations (byte addresses), stage 0
    const uint32_t dA0 = (uint32_t)(uintptr_t)(&As[0]) +
                         (uint32_t)(r_st * LDS_PITCH + c_st) * 2;
    const uint32_t dB0 = (uint32_t)(uintptr_t)(&Bs[0]) +
                         (uint32_t)(r_st * LDS_PITCH + c_st) * 2;

    // per-lane fragment base indices (halves, stage 0)
    const int aIdx = (wm + lm) * LDS_PITCH + kfA;
    const int bIdx = (wn + lm) * LDS_PITCH + kfB;

    // prologue: fill stage 0
    async_copy_b128(dA0,      gA);
    async_copy_b128(dA0 + 32, gA + 16);
    async_copy_b128(dB0,      gB);
    async_copy_b128(dB0 + 32, gB + 16);
    gA += BK; gB += BK;
    wait_async0();
    __syncthreads();

    uint32_t so = 0;                        // stage offset in halves {0, STAGE_H}

#pragma unroll 1
    for (int kt = 0; kt < KSTEPS; ++kt) {
        const uint32_t soN = so ^ (uint32_t)STAGE_H;

        // issue async DMA for next stage; overlaps with WMMA below
        if (kt + 1 < KSTEPS) {
            const uint32_t dA = dA0 + soN * 2;
            const uint32_t dB = dB0 + soN * 2;
            async_copy_b128(dA,      gA);
            async_copy_b128(dA + 32, gA + 16);
            async_copy_b128(dB,      gB);
            async_copy_b128(dB + 32, gB + 16);
            gA += BK; gB += BK;
        }

        const _Float16* Ac = &As[so + aIdx];
        const _Float16* Bc = &Bs[so + bIdx];

        // ---- A fragments (ISA 16-bit A lane layout) ----
        const v16h a0 = concat16(*(const v8h*)(Ac),
                                 *(const v8h*)(Ac + 16));
        const v16h a1 = concat16(*(const v8h*)(Ac + 16 * LDS_PITCH),
                                 *(const v8h*)(Ac + 16 * LDS_PITCH + 16));

        // ---- per-j B fragment + 2 WMMAs ----
#pragma unroll
        for (int j = 0; j < 4; ++j) {
            const _Float16* p = Bc + j * (16 * LDS_PITCH);
            const v16h b = concat16(*(const v8h*)(p), *(const v8h*)(p + 16));

            acc[0][j] = __builtin_amdgcn_wmma_f32_16x16x32_f16(
                false, a0, false, b, (short)0, acc[0][j], false, false);
            acc[1][j] = __builtin_amdgcn_wmma_f32_16x16x32_f16(
                false, a1, false, b, (short)0, acc[1][j], false, false);
        }

        wait_async0();      // next stage fully in LDS
        __syncthreads();    // visible to all waves before next iteration
        so = soN;
    }

    // ---- epilogue: VGPR v holds M=v (lanes 0-15) / M=v+8 (lanes 16-31) ----
    const int rbase = m0 + wm + ((lane < 16) ? 0 : 8);
    const int cbase = n0 + wn + lm;
#pragma unroll
    for (int i = 0; i < 2; ++i)
#pragma unroll
        for (int j = 0; j < 4; ++j)
#pragma unroll
            for (int v = 0; v < 8; ++v)
                C[(size_t)(rbase + i * 16 + v) * GEMM_N + cbase + j * 16] =
                    acc[i][j][v];
}

// ---------------------------------------------------------------------------
extern "C" void kernel_launch(void* const* d_in, const int* in_sizes, int n_in,
                              void* d_out, int out_size, void* d_ws, size_t ws_size,
                              hipStream_t stream) {
    (void)in_sizes; (void)n_in; (void)out_size;

    const float* x = (const float*)d_in[0];                 // [8192,512] f32
    const float* w = (const float*)d_in[1];                 // [512,512,8] f32
    float*       c = (float*)d_out;                         // [8192,512] f32

    _Float16* Aws = (_Float16*)d_ws;                        // 64 MB
    _Float16* Bws = (_Float16*)((char*)d_ws + (size_t)8192 * 4096 * 2);
    (void)ws_size;  // requires >= 71,303,168 bytes

    kan_basis_f16<<<(8192 * 512) / 256, 256, 0, stream>>>(x, Aws);
    kan_coeff_f16<<<(512 * 512) / 256, 256, 0, stream>>>(w, Bws);

    dim3 grid(GEMM_N / BN, 8192 / BM);                      // (4, 64)
    kan_wmma_gemm<<<grid, 256, 0, stream>>>(Aws, Bws, c);
}
